// MoleculeGCN_21560735826432
// MI455X (gfx1250) — compile-verified, matrix operand order checked by reading, not run on hardware
//
#include <hip/hip_runtime.h>

typedef __attribute__((ext_vector_type(16))) __bf16 v16bf;
typedef __attribute__((ext_vector_type(8)))  __bf16 v8bf;
typedef __attribute__((ext_vector_type(8)))  float  v8f;

#define FEAT 64   // output feature width of both GCN layers

// ---------------- degree / normalization ----------------
__global__ void k_deg_init(float* deg, int n) {
  int i = blockIdx.x * blockDim.x + threadIdx.x;
  if (i < n) deg[i] = 1.0f;                       // self-loop contributes 1
}

__global__ void k_deg_count(const long long* __restrict__ dst, float* deg, int nE) {
  int e = blockIdx.x * blockDim.x + threadIdx.x;
  if (e < nE) atomicAdd(&deg[(int)dst[e]], 1.0f);
}

__global__ void k_rsqrt_inplace(float* deg, int n) {
  int i = blockIdx.x * blockDim.x + threadIdx.x;
  if (i < n) deg[i] = rsqrtf(deg[i]);             // deg >= 1 always
}

// ---------------- conversions / packing ----------------
__global__ void k_f32_to_bf16(const float* __restrict__ src, __bf16* __restrict__ dst,
                              long long total) {
  long long i = (long long)blockIdx.x * blockDim.x + threadIdx.x;
  long long stride = (long long)gridDim.x * blockDim.x;
  for (; i < total; i += stride) dst[i] = (__bf16)src[i];
}

// Pack W [K x 64] (f32 row-major) into the per-lane bf16 WMMA B-matrix layout.
// Wp index = ((nt*nk + kc)*32 + lane)*16 + j, nk = K/32, nt = N-tile (4 of 16 cols).
// Per 16x16x32 B tile: lane holds column n = lane&15; kbase = (lane>>4)*8;
//   elems 0..7  = K rows  kbase..kbase+7
//   elems 8..15 = K rows  16+kbase..16+kbase+7
__global__ void k_pack_w(const float* __restrict__ W, __bf16* __restrict__ Wp, int K) {
  int nk = K >> 5;
  int total = 4 * nk * 32 * 16;
  int i = blockIdx.x * blockDim.x + threadIdx.x;
  int stride = gridDim.x * blockDim.x;
  for (; i < total; i += stride) {
    int j     = i & 15;
    int lane  = (i >> 4) & 31;
    int rest  = i >> 9;
    int kc    = rest % nk;
    int nt    = rest / nk;
    int ncol  = lane & 15;
    int kbase = (lane >> 4) * 8;
    int k = kc * 32 + ((j < 8) ? (kbase + j) : (16 + kbase + (j - 8)));
    Wp[i] = (__bf16)W[k * FEAT + nt * 16 + ncol];
  }
}

// ---------------- WMMA GEMM:  H[n x 64] = A[n x K](bf16) @ Wp(packed bf16) ----------------
// One wave per 16-row M-tile; 4 accumulator tiles cover all 64 output columns,
// so the A row block is loaded exactly once per wave.
__global__ __launch_bounds__(256) void k_gemm_bf16(const __bf16* __restrict__ A,
                                                   const __bf16* __restrict__ Wp,
                                                   float* __restrict__ H,
                                                   int n, int K) {
  const int lane   = threadIdx.x & 31;
  const int wave   = threadIdx.x >> 5;
  const int mtile  = blockIdx.x * 8 + wave;
  const int mtiles = (n + 15) >> 4;
  if (mtile >= mtiles) return;                    // wave-uniform exit

  const int nk = K >> 5;
  int mrow = mtile * 16 + (lane & 15);
  if (mrow > n - 1) mrow = n - 1;                 // clamp: EXEC stays all-ones for WMMA
  const int kbase = (lane >> 4) * 8;

  v8f acc[4] = {v8f{}, v8f{}, v8f{}, v8f{}};
  const __bf16* arow = A + (long long)mrow * K;

  for (int kc = 0; kc < nk; ++kc) {
    // A-matrix lane fragment: two contiguous 16-byte loads
    v8bf lo = *(const v8bf*)(arow + kc * 32 + kbase);
    v8bf hi = *(const v8bf*)(arow + kc * 32 + 16 + kbase);
    v16bf a;
#pragma unroll
    for (int j = 0; j < 8; ++j) { a[j] = lo[j]; a[j + 8] = hi[j]; }

#pragma unroll
    for (int nt = 0; nt < 4; ++nt) {
      v16bf b = *(const v16bf*)(Wp + (((long long)nt * nk + kc) * 32 + lane) * 16);
      acc[nt] = __builtin_amdgcn_wmma_f32_16x16x32_bf16(
          /*neg_a=*/false, a, /*neg_b=*/false, b,
          /*c_mod=*/(short)0, acc[nt], /*reuse_a=*/false, /*reuse_b=*/false);
    }
  }

  // C/D layout: VGPR j -> lanes 0-15: (M=j, N=lane); lanes 16-31: (M=8+j, N=lane-16)
  const int rbase = mtile * 16 + (lane >> 4) * 8;
  const int col   = lane & 15;
#pragma unroll
  for (int nt = 0; nt < 4; ++nt) {
#pragma unroll
    for (int j = 0; j < 8; ++j) {
      int row = rbase + j;
      if (row < n) H[(long long)row * FEAT + nt * 16 + col] = acc[nt][j];
    }
  }
}

// ---------------- bias init / edge scatter / finalize ----------------
__global__ void k_bias_init(float* __restrict__ acc, const float* __restrict__ b,
                            long long total) {
  long long i = (long long)blockIdx.x * blockDim.x + threadIdx.x;
  long long stride = (long long)gridDim.x * blockDim.x;
  for (; i < total; i += stride) acc[i] = b[i & (FEAT - 1)];
}

// 256 threads = 4 edges x 64 features; edge metadata staged through LDS.
__global__ __launch_bounds__(256) void k_scatter(const float* __restrict__ h,
                                                 const float* __restrict__ dinv,
                                                 const long long* __restrict__ src,
                                                 const long long* __restrict__ dst,
                                                 float* __restrict__ acc, int nE) {
  __shared__ int   s_s[4];
  __shared__ int   s_d[4];
  __shared__ float s_w[4];
  const long long base = (long long)blockIdx.x * 4;
  const int t = threadIdx.x;
  if (t < 4) {
    long long e = base + t;
    if (e < nE) {
      int s = (int)src[e];
      int d = (int)dst[e];
      s_s[t] = s; s_d[t] = d; s_w[t] = dinv[s] * dinv[d];
    }
  }
  __syncthreads();
  const int sub = t >> 6;
  long long e = base + sub;
  if (e < nE) {
    int f = t & (FEAT - 1);
    float v = h[(long long)s_s[sub] * FEAT + f] * s_w[sub];
    atomicAdd(acc + (long long)s_d[sub] * FEAT + f, v);
  }
}

// out = acc + h[i]*dinv[i]^2 (self-loop term), optional ReLU; emit f32 and/or bf16
__global__ void k_finalize(const float* __restrict__ acc, const float* __restrict__ h,
                           const float* __restrict__ dinv,
                           float* __restrict__ outf, __bf16* __restrict__ outb,
                           long long total, int do_relu) {
  long long i = (long long)blockIdx.x * blockDim.x + threadIdx.x;
  long long stride = (long long)gridDim.x * blockDim.x;
  for (; i < total; i += stride) {
    int node = (int)(i >> 6);
    float di = dinv[node];
    float v = acc[i] + h[i] * di * di;
    if (do_relu) v = fmaxf(v, 0.0f);
    if (outf) outf[i] = v;
    if (outb) outb[i] = (__bf16)v;
  }
}

// ---------------- launcher ----------------
extern "C" void kernel_launch(void* const* d_in, const int* in_sizes, int n_in,
                              void* d_out, int out_size, void* d_ws, size_t ws_size,
                              hipStream_t stream) {
  const float*     x   = (const float*)d_in[0];
  const long long* ei  = (const long long*)d_in[1];   // int64 [2, E]
  const float*     W1  = (const float*)d_in[2];
  const float*     b1  = (const float*)d_in[3];
  const float*     W2  = (const float*)d_in[4];
  const float*     b2  = (const float*)d_in[5];

  const int N = in_sizes[0] / 128;
  const int E = in_sizes[1] / 2;
  const long long* srcp = ei;
  const long long* dstp = ei + E;
  float* outp = (float*)d_out;

  auto align256 = [](size_t v) { return (v + 255) & ~(size_t)255; };
  char* p = (char*)d_ws;
  float*  dinv = (float*)p;  p += align256((size_t)N * 4);
  __bf16* xb   = (__bf16*)p; p += align256((size_t)N * 128 * 2);
  __bf16* w1p  = (__bf16*)p; p += align256((size_t)4 * 4 * 32 * 16 * 2);
  __bf16* w2p  = (__bf16*)p; p += align256((size_t)4 * 2 * 32 * 16 * 2);
  float*  h1   = (float*)p;  p += align256((size_t)N * FEAT * 4);
  float*  acc1 = (float*)p;  p += align256((size_t)N * FEAT * 4);
  __bf16* o1b  = (__bf16*)p; p += align256((size_t)N * FEAT * 2);
  float*  h2   = (float*)p;  p += align256((size_t)N * FEAT * 4);

  const long long NF = (long long)N * FEAT;
  const int mtiles = (N + 15) / 16;

  // degrees -> dinv (in place)
  k_deg_init<<<(N + 255) / 256, 256, 0, stream>>>(dinv, N);
  k_deg_count<<<(E + 255) / 256, 256, 0, stream>>>(dstp, dinv, E);
  k_rsqrt_inplace<<<(N + 255) / 256, 256, 0, stream>>>(dinv, N);

  // precision/packing prep
  k_f32_to_bf16<<<2048, 256, 0, stream>>>(x, xb, (long long)N * 128);
  k_pack_w<<<32, 256, 0, stream>>>(W1, w1p, 128);
  k_pack_w<<<16, 256, 0, stream>>>(W2, w2p, 64);

  // ---- layer 1 ----
  k_gemm_bf16<<<(mtiles + 7) / 8, 256, 0, stream>>>(xb, w1p, h1, N, 128);
  k_bias_init<<<2048, 256, 0, stream>>>(acc1, b1, NF);
  k_scatter<<<(E + 3) / 4, 256, 0, stream>>>(h1, dinv, srcp, dstp, acc1, E);
  k_finalize<<<2048, 256, 0, stream>>>(acc1, h1, dinv, (float*)nullptr, o1b, NF, 1);

  // ---- layer 2 (accumulate straight into d_out) ----
  k_gemm_bf16<<<(mtiles + 7) / 8, 256, 0, stream>>>(o1b, w2p, h2, N, 64);
  k_bias_init<<<2048, 256, 0, stream>>>(outp, b2, NF);
  k_scatter<<<(E + 3) / 4, 256, 0, stream>>>(h2, dinv, srcp, dstp, outp, E);
  k_finalize<<<2048, 256, 0, stream>>>(outp, h2, dinv, outp, (__bf16*)nullptr, NF, 0);
}
